// TransformerBlock_73375221284966
// MI455X (gfx1250) — matrix operand order repaired and checked by reading, not compiled, not run on hardware
//
#include <hip/hip_runtime.h>
#include <math.h>

// ---------------------------------------------------------------------------
// TransformerBlock for MI455X (gfx1250, wave32, WMMA + async global->LDS).
// B=8, N=1024, C=384, H=6 (hd=64), E=4, band half-width 3.
//
// GEMM: block tile 128x128 (8 waves), wave tile 32x64 (2 A-frags x 4 B-frags
// = 8 v_wmma_f32_16x16x32_f16 per K-step), double-buffered LDS staged with
// GLOBAL_LOAD_ASYNC_TO_LDS_B128 (ASYNCcnt) when the builtin is available.
// ---------------------------------------------------------------------------

#define TOKENS   8192
#define CDIM     384
#define HEADS    6
#define HDIM     64
#define NEXP     4
#define KSTR     40   // LDS K-stride in halves: 80B, 16B-aligned, conflict-free

typedef __attribute__((ext_vector_type(16))) _Float16 v16h;
typedef __attribute__((ext_vector_type(8)))  float    v8f;

union Frag16 { v16h h; uint4 q[2]; };

#if defined(__gfx1250__) && \
    __has_builtin(__builtin_amdgcn_global_load_async_to_lds_b128) && \
    __has_builtin(__builtin_amdgcn_s_wait_asynccnt)
#define HAVE_ASYNC_LDS 1
#else
#define HAVE_ASYNC_LDS 0
#endif

#if HAVE_ASYNC_LDS
// Builtin signature (from hipcc diagnostic): first param is
// 'int __attribute__((vector_size(16)))*' in the global (AS1) address space;
// second is the LDS (AS3) counterpart.
typedef int v4i_g __attribute__((vector_size(16)));
typedef __attribute__((address_space(1))) v4i_g* as1_v4i;
typedef __attribute__((address_space(3))) v4i_g* as3_v4i;
#endif

// 16-byte global->LDS copy. Async path: direct-to-LDS DMA tracked by ASYNCcnt
// (no VGPR round trip). Pointer casts go through integers: AS1 ptr == 64-bit
// VA; AS3 ptr == low 32 bits of the generic LDS address (aperture rule).
__device__ __forceinline__ void cp16(const _Float16* g, _Float16* l) {
#if HAVE_ASYNC_LDS
  __builtin_amdgcn_global_load_async_to_lds_b128(
      (as1_v4i)(uintptr_t)g,
      (as3_v4i)(uint32_t)(uintptr_t)l,
      0, 0);
#else
  *(uint4*)l = *(const uint4*)g;
#endif
}

__device__ __forceinline__ float waveAllSum(float v) {
  #pragma unroll
  for (int m = 16; m >= 1; m >>= 1) v += __shfl_xor(v, m, 32);
  return v;
}

// ---------------- weight transpose + f32->f16 convert ----------------------
__global__ void transpose_to_f16_kernel(const float* __restrict__ in,
                                        _Float16* __restrict__ out,
                                        int R, int Ccols) {
  int i = blockIdx.x * 256 + threadIdx.x;
  if (i < R * Ccols) {
    int r = i / Ccols, c = i % Ccols;
    out[(size_t)c * R + r] = (_Float16)in[(size_t)r * Ccols + c];
  }
}

// ---------------- LayerNorm1 -> f16 ----------------------------------------
__global__ void ln1_kernel(const float* __restrict__ x,
                           const float* __restrict__ w,
                           const float* __restrict__ b,
                           _Float16* __restrict__ out) {
  int wave = threadIdx.x >> 5, lane = threadIdx.x & 31;
  int token = blockIdx.x * 8 + wave;
  const float* row = x + (size_t)token * CDIM;
  float vals[12];
  float s = 0.f;
  #pragma unroll
  for (int i = 0; i < 12; ++i) { vals[i] = row[lane + i * 32]; s += vals[i]; }
  float mean = waveAllSum(s) * (1.0f / CDIM);
  float vs = 0.f;
  #pragma unroll
  for (int i = 0; i < 12; ++i) { float d = vals[i] - mean; vs += d * d; }
  float inv = rsqrtf(waveAllSum(vs) * (1.0f / CDIM) + 1e-5f);
  #pragma unroll
  for (int i = 0; i < 12; ++i) {
    int c = lane + i * 32;
    out[(size_t)token * CDIM + c] = (_Float16)((vals[i] - mean) * inv * w[c] + b[c]);
  }
}

// ---------------- generic f16 WMMA GEMM with fused epilogues ----------------
// C[M,N] = A[M,K] * W[K,N]; W supplied pre-transposed as Bt[N][K] (f16).
// Block: 256 threads = 8 waves; block tile 128x128; wave tile 32x64.
// mode 0: out32 = acc (+bias)
// mode 1: out32 = res + scalePtr[0] * (acc + bias)
// mode 2: out16 = gelu_exact(acc + bias)
// mode 3: out32 += rowW[row*4] * (acc + bias)
__global__ void __launch_bounds__(256)
gemm_f16_kernel(const _Float16* __restrict__ A, const _Float16* __restrict__ Bt,
                int M, int N, int K,
                const float* __restrict__ bias,
                float* __restrict__ out32, _Float16* __restrict__ out16,
                const float* __restrict__ res, const float* __restrict__ scalePtr,
                const float* __restrict__ rowW, int mode) {
  __shared__ _Float16 ldsA[2][128 * KSTR];
  __shared__ _Float16 ldsB[2][128 * KSTR];
  int tid  = threadIdx.x;
  int wave = tid >> 5, lane = tid & 31;
  int waveM = wave >> 1;   // 0..3 -> 32-row strip
  int waveN = wave & 1;    // 0..1 -> 64-col strip
  int rowBlock = blockIdx.y * 128;
  int colBlock = blockIdx.x * 128;

  v8f acc[2][4] = {};

  // staging assignments: 256 threads, 16 halves (2x16B) each per matrix
  int sr = tid >> 1;            // 0..127 row (A) / col (B)
  int sk = (tid & 1) * 16;      // K offset 0 or 16

  const int l15 = lane & 15;
  const int lhi = lane >> 4;
  const int khb = lhi * 8;      // A frag: lanes 0-15 K 0-7/16-23, 16-31 K 8-15/24-31
  const int kg  = lhi * 16;     // (unused) B frag uses khb: same K order as A frag

  const _Float16* aSrc = A  + (size_t)(rowBlock + sr) * K + sk;
  const _Float16* bSrc = Bt + (size_t)(colBlock + sr) * K + sk;
  _Float16* aDst = &ldsA[0][0] + sr * KSTR + sk;
  _Float16* bDst = &ldsB[0][0] + sr * KSTR + sk;
  const int bufStride = 128 * KSTR;

  // prologue: stage K-step 0 into buffer 0 (4 ops/thread)
  cp16(aSrc,     aDst);
  cp16(aSrc + 8, aDst + 8);
  cp16(bSrc,     bDst);
  cp16(bSrc + 8, bDst + 8);

  int nsteps = K >> 5;
  for (int s = 0; s < nsteps; ++s) {
    int cur = s & 1;
    if (s + 1 < nsteps) {
      // stage next K-step into the other buffer while this one computes
      int kb = (s + 1) << 5;
      int nb = cur ^ 1;
      cp16(aSrc + kb,     aDst + nb * bufStride);
      cp16(aSrc + kb + 8, aDst + nb * bufStride + 8);
      cp16(bSrc + kb,     bDst + nb * bufStride);
      cp16(bSrc + kb + 8, bDst + nb * bufStride + 8);
#if HAVE_ASYNC_LDS
      __builtin_amdgcn_s_wait_asynccnt(4);   // current buffer done, next in flight
#endif
    } else {
#if HAVE_ASYNC_LDS
      __builtin_amdgcn_s_wait_asynccnt(0);
#endif
    }
    __syncthreads();

    const _Float16* la = &ldsA[cur][0];
    const _Float16* lb = &ldsB[cur][0];

    Frag16 af[2];
    #pragma unroll
    for (int m = 0; m < 2; ++m) {
      int arow = waveM * 32 + m * 16 + l15;
      af[m].q[0] = *(const uint4*)&la[arow * KSTR + khb];
      af[m].q[1] = *(const uint4*)&la[arow * KSTR + 16 + khb];
    }
    #pragma unroll
    for (int t = 0; t < 4; ++t) {
      Frag16 bf;
      int bcol = waveN * 64 + t * 16 + l15;
      bf.q[0] = *(const uint4*)&lb[bcol * KSTR + khb];
      bf.q[1] = *(const uint4*)&lb[bcol * KSTR + 16 + khb];
      acc[0][t] = __builtin_amdgcn_wmma_f32_16x16x32_f16(
          false, af[0].h, false, bf.h, (short)0, acc[0][t], false, false);
      acc[1][t] = __builtin_amdgcn_wmma_f32_16x16x32_f16(
          false, af[1].h, false, bf.h, (short)0, acc[1][t], false, false);
    }
    __syncthreads();
  }

  // epilogue: C/D layout -> lane 0-15: N=lane, M=r ; lane 16-31: N=lane-16, M=r+8
  int colL0 = waveN * 64 + l15;
  #pragma unroll
  for (int m = 0; m < 2; ++m) {
    int rowL = waveM * 32 + m * 16 + 8 * lhi;
    #pragma unroll
    for (int t = 0; t < 4; ++t) {
      int col = colBlock + colL0 + t * 16;
      float bv = bias ? bias[col] : 0.0f;
      #pragma unroll
      for (int r = 0; r < 8; ++r) {
        int row = rowBlock + rowL + r;
        size_t idx = (size_t)row * N + col;
        float v = acc[m][t][r] + bv;
        if (mode == 0) {
          out32[idx] = v;
        } else if (mode == 1) {
          out32[idx] = res[idx] + scalePtr[0] * v;
        } else if (mode == 2) {
          out16[idx] = (_Float16)(0.5f * v * (1.0f + erff(v * 0.70710678118f)));
        } else {
          out32[idx] += rowW[(size_t)row * NEXP] * v;
        }
      }
    }
  }
}

// ---------------- banded attention (window 7), wave per (b,h,n) -------------
__global__ void attn_kernel(const float* __restrict__ qkv,
                            const float* __restrict__ temperature,
                            _Float16* __restrict__ o) {
  int wave = threadIdx.x >> 5, lane = threadIdx.x & 31;
  int g = blockIdx.x * 8 + wave;           // 0 .. 8*6*1024-1
  int b = g / (HEADS * 1024);
  int rem = g % (HEADS * 1024);
  int h = rem / 1024;
  int n = rem % 1024;
  size_t tok = (size_t)b * 1024 + n;

  const float* qp = qkv + tok * (3 * CDIM) + h * HDIM;
  float q0 = qp[lane], q1 = qp[lane + 32];
  float invt = 0.125f / temperature[h];    // hd^-0.5 / temp

  float sc[7];
  float mx = -1e30f;
  #pragma unroll
  for (int j = 0; j < 7; ++j) {
    int m = n + j - 3;
    if (m >= 0 && m < 1024) {
      const float* kp = qkv + ((size_t)b * 1024 + m) * (3 * CDIM) + CDIM + h * HDIM;
      float d = q0 * kp[lane] + q1 * kp[lane + 32];
      d = waveAllSum(d) * invt;
      sc[j] = d;
      mx = fmaxf(mx, d);
    } else {
      sc[j] = -1e30f;
    }
  }
  float p[7], se = 0.f;
  #pragma unroll
  for (int j = 0; j < 7; ++j) {
    p[j] = (sc[j] > -1e29f) ? __expf(sc[j] - mx) : 0.0f;
    se += p[j];
  }
  float o0 = 0.f, o1 = 0.f;
  #pragma unroll
  for (int j = 0; j < 7; ++j) {
    int m = n + j - 3;
    if (p[j] != 0.0f) {
      const float* vp = qkv + ((size_t)b * 1024 + m) * (3 * CDIM) + 2 * CDIM + h * HDIM;
      o0 += p[j] * vp[lane];
      o1 += p[j] * vp[lane + 32];
    }
  }
  float inv = 1.0f / se;
  o[tok * CDIM + h * HDIM + lane]      = (_Float16)(o0 * inv);
  o[tok * CDIM + h * HDIM + lane + 32] = (_Float16)(o1 * inv);
}

// ---------------- LN2 + gate softmax + top2 ---------------------------------
__global__ void ln2_gate_kernel(const float* __restrict__ x1,
                                const float* __restrict__ w,
                                const float* __restrict__ b,
                                const float* __restrict__ gw,
                                const float* __restrict__ gb,
                                _Float16* __restrict__ h2,
                                float* __restrict__ comb) {
  int wave = threadIdx.x >> 5, lane = threadIdx.x & 31;
  int token = blockIdx.x * 8 + wave;
  const float* row = x1 + (size_t)token * CDIM;
  float vals[12];
  float s = 0.f;
  #pragma unroll
  for (int i = 0; i < 12; ++i) { vals[i] = row[lane + i * 32]; s += vals[i]; }
  float mean = waveAllSum(s) * (1.0f / CDIM);
  float vs = 0.f;
  #pragma unroll
  for (int i = 0; i < 12; ++i) { float d = vals[i] - mean; vs += d * d; }
  float inv = rsqrtf(waveAllSum(vs) * (1.0f / CDIM) + 1e-5f);
  float nrm[12];
  #pragma unroll
  for (int i = 0; i < 12; ++i) {
    int c = lane + i * 32;
    nrm[i] = (vals[i] - mean) * inv * w[c] + b[c];
    h2[(size_t)token * CDIM + c] = (_Float16)nrm[i];
  }
  float logit[NEXP];
  #pragma unroll
  for (int e = 0; e < NEXP; ++e) {
    float pacc = 0.f;
    #pragma unroll
    for (int i = 0; i < 12; ++i) pacc += nrm[i] * gw[(size_t)(lane + i * 32) * NEXP + e];
    logit[e] = waveAllSum(pacc) + gb[e];
  }
  float mx = logit[0];
  #pragma unroll
  for (int e = 1; e < NEXP; ++e) mx = fmaxf(mx, logit[e]);
  float pe[NEXP], se = 0.f;
  #pragma unroll
  for (int e = 0; e < NEXP; ++e) { pe[e] = __expf(logit[e] - mx); se += pe[e]; }
  #pragma unroll
  for (int e = 0; e < NEXP; ++e) pe[e] /= se;
  int i1 = 0;
  #pragma unroll
  for (int e = 1; e < NEXP; ++e) if (pe[e] > pe[i1]) i1 = e;
  int i2 = (i1 == 0) ? 1 : 0;
  #pragma unroll
  for (int e = 0; e < NEXP; ++e) if (e != i1 && pe[e] > pe[i2]) i2 = e;
  float g1 = pe[i1], g2 = pe[i2], gs = g1 + g2;
  if (lane < NEXP) {
    float v = (lane == i1) ? g1 / gs : ((lane == i2) ? g2 / gs : 0.0f);
    comb[(size_t)token * NEXP + lane] = v;
  }
}

// ---------------- elementwise helpers ---------------------------------------
__global__ void zero_kernel(float* __restrict__ p, int n) {
  int i = blockIdx.x * 256 + threadIdx.x;
  if (i < n) p[i] = 0.0f;
}

__global__ void final_add_kernel(const float* __restrict__ x1,
                                 const float* __restrict__ accum,
                                 const float* __restrict__ mlpScale,
                                 float* __restrict__ out, int n) {
  int i = blockIdx.x * 256 + threadIdx.x;
  if (i < n) out[i] = x1[i] + mlpScale[0] * accum[i];
}

// ---------------------------------------------------------------------------
extern "C" void kernel_launch(void* const* d_in, const int* in_sizes, int n_in,
                              void* d_out, int out_size, void* d_ws, size_t ws_size,
                              hipStream_t stream) {
  const float* x         = (const float*)d_in[0];
  const float* ln1_w     = (const float*)d_in[1];
  const float* ln1_b     = (const float*)d_in[2];
  const float* qkv_w     = (const float*)d_in[3];
  const float* proj_w    = (const float*)d_in[4];
  const float* proj_b    = (const float*)d_in[5];
  const float* temp      = (const float*)d_in[6];
  const float* ln2_w     = (const float*)d_in[7];
  const float* ln2_b     = (const float*)d_in[8];
  const float* gate_w    = (const float*)d_in[9];
  const float* gate_b    = (const float*)d_in[10];
  const float* ew1       = (const float*)d_in[11];
  const float* eb1       = (const float*)d_in[12];
  const float* ew2       = (const float*)d_in[13];
  const float* eb2       = (const float*)d_in[14];
  const float* attn_scale= (const float*)d_in[15];
  const float* mlp_scale = (const float*)d_in[16];
  float* out = (float*)d_out;

  // workspace layout (all offsets multiples of 256B)
  char* ws = (char*)d_ws;
  size_t off = 0;
  auto alloc = [&](size_t bytes) { char* p = ws + off; off += (bytes + 255) & ~size_t(255); return p; };
  _Float16* h1    = (_Float16*)alloc((size_t)TOKENS * CDIM * 2);
  _Float16* qkvwT = (_Float16*)alloc((size_t)3 * CDIM * CDIM * 2);
  float*    qkv   = (float*)   alloc((size_t)TOKENS * 3 * CDIM * 4);
  _Float16* o16   = (_Float16*)alloc((size_t)TOKENS * CDIM * 2);
  _Float16* projwT= (_Float16*)alloc((size_t)CDIM * CDIM * 2);
  float*    x1    = (float*)   alloc((size_t)TOKENS * CDIM * 4);
  _Float16* h2    = (_Float16*)alloc((size_t)TOKENS * CDIM * 2);
  float*    comb  = (float*)   alloc((size_t)TOKENS * NEXP * 4);
  _Float16* ew1T  = (_Float16*)alloc((size_t)NEXP * 2 * CDIM * CDIM * 2);
  _Float16* ew2T  = (_Float16*)alloc((size_t)NEXP * 2 * CDIM * CDIM * 2);
  _Float16* hid   = (_Float16*)alloc((size_t)TOKENS * 2 * CDIM * 2);
  float*    accum = (float*)   alloc((size_t)TOKENS * CDIM * 4);

  // 0) weight transpose + convert  (Bt layout: [N][K])
  {
    int n = CDIM * 3 * CDIM;
    transpose_to_f16_kernel<<<(n + 255) / 256, 256, 0, stream>>>(qkv_w, qkvwT, CDIM, 3 * CDIM);
    n = CDIM * CDIM;
    transpose_to_f16_kernel<<<(n + 255) / 256, 256, 0, stream>>>(proj_w, projwT, CDIM, CDIM);
    for (int e = 0; e < NEXP; ++e) {
      n = CDIM * 2 * CDIM;
      transpose_to_f16_kernel<<<(n + 255) / 256, 256, 0, stream>>>(
          ew1 + (size_t)e * CDIM * 2 * CDIM, ew1T + (size_t)e * 2 * CDIM * CDIM, CDIM, 2 * CDIM);
      transpose_to_f16_kernel<<<(n + 255) / 256, 256, 0, stream>>>(
          ew2 + (size_t)e * 2 * CDIM * CDIM, ew2T + (size_t)e * CDIM * 2 * CDIM, 2 * CDIM, CDIM);
    }
  }

  // 1) LN1
  ln1_kernel<<<TOKENS / 8, 256, 0, stream>>>(x, ln1_w, ln1_b, h1);

  // 2) QKV GEMM (mode 0, no bias)
  {
    dim3 grid((3 * CDIM) / 128, TOKENS / 128);
    gemm_f16_kernel<<<grid, 256, 0, stream>>>(h1, qkvwT, TOKENS, 3 * CDIM, CDIM,
        nullptr, qkv, nullptr, nullptr, nullptr, nullptr, 0);
  }

  // 3) banded attention
  attn_kernel<<<(TOKENS * HEADS) / 8, 256, 0, stream>>>(qkv, temp, o16);

  // 4) proj GEMM + residual (mode 1)
  {
    dim3 grid(CDIM / 128, TOKENS / 128);
    gemm_f16_kernel<<<grid, 256, 0, stream>>>(o16, projwT, TOKENS, CDIM, CDIM,
        proj_b, x1, nullptr, x, attn_scale, nullptr, 1);
  }

  // 5) LN2 + gate + top2
  ln2_gate_kernel<<<TOKENS / 8, 256, 0, stream>>>(x1, ln2_w, ln2_b, gate_w, gate_b, h2, comb);

  // 6) zero MoE accumulator
  {
    int n = TOKENS * CDIM;
    zero_kernel<<<(n + 255) / 256, 256, 0, stream>>>(accum, n);
  }

  // 7) experts: fc1 (GELU, f16 out) then fc2 (gate-weighted accumulate)
  for (int e = 0; e < NEXP; ++e) {
    dim3 g1((2 * CDIM) / 128, TOKENS / 128);
    gemm_f16_kernel<<<g1, 256, 0, stream>>>(h2, ew1T + (size_t)e * 2 * CDIM * CDIM,
        TOKENS, 2 * CDIM, CDIM,
        eb1 + (size_t)e * 2 * CDIM, nullptr, hid, nullptr, nullptr, nullptr, 2);
    dim3 g2(CDIM / 128, TOKENS / 128);
    gemm_f16_kernel<<<g2, 256, 0, stream>>>(hid, ew2T + (size_t)e * CDIM * 2 * CDIM,
        TOKENS, CDIM, 2 * CDIM,
        eb2 + (size_t)e * CDIM, accum, nullptr, nullptr, nullptr, comb + e, 3);
  }

  // 8) final residual
  {
    int n = TOKENS * CDIM;
    final_add_kernel<<<(n + 255) / 256, 256, 0, stream>>>(x1, accum, mlp_scale, out, n);
  }

  (void)in_sizes; (void)n_in; (void)out_size; (void)ws_size;
}
